// DifferentialAttention_28235115004581
// MI455X (gfx1250) — matrix operand order repaired and checked by reading, not compiled, not run on hardware
//
#include <hip/hip_runtime.h>

// ---------------------------------------------------------------------------
// Differential attention for MI455X (gfx1250, wave32, WMMA 16x16x32 bf16).
// Pipeline: lambda -> cvt/transposes -> 3x QKV GEMM (bf16 WMMA, async/TDM
// double-buffered LDS staging) -> fused flash diff-attention -> out GEMM.
// Workspace usage ~82 MB.
// ---------------------------------------------------------------------------

typedef __bf16 bf16_t;
typedef __attribute__((ext_vector_type(16))) __bf16 v16bf;
typedef __attribute__((ext_vector_type(4)))  __bf16 v4bf;
typedef __attribute__((ext_vector_type(8)))  float  v8f;

// Types matching the gfx1250 builtin prototypes (gcc vector style).
typedef int          async_b128_t __attribute__((__vector_size__(16)));
typedef unsigned int tdm_u32x4    __attribute__((__vector_size__(16)));
typedef int          tdm_i32x4    __attribute__((__vector_size__(16)));
typedef int          tdm_i32x8    __attribute__((__vector_size__(32)));

#define SEQ     2048
#define NHEADS  16
#define DHEAD   64
#define BS      2

#define ASG __attribute__((address_space(1)))
#define ASL __attribute__((address_space(3)))

// CDNA5 async global->LDS copy (ASYNCcnt) -- guarded, falls back to sync path.
#if defined(__gfx1250__) && __has_builtin(__builtin_amdgcn_global_load_async_to_lds_b128) && \
    __has_builtin(__builtin_amdgcn_s_wait_asynccnt)
#define USE_ASYNC_COPY 1
#else
#define USE_ASYNC_COPY 0
#endif

// CDNA5 Tensor Data Mover (TENSORcnt) for the GEMM B tile -- guarded.
#if defined(__gfx1250__) && __has_builtin(__builtin_amdgcn_tensor_load_to_lds) && \
    __has_builtin(__builtin_amdgcn_s_wait_tensorcnt)
#define USE_TDM 1
#else
#define USE_TDM 0
#endif

#if USE_TDM
#define ASYNC_PER_THREAD 2   // only A staged per-thread
#else
#define ASYNC_PER_THREAD 4   // A + B staged per-thread
#endif

__device__ __forceinline__ ASG async_b128_t* to_global_b128(const void* p) {
  return (ASG async_b128_t*)(unsigned long long)(size_t)p;
}
__device__ __forceinline__ ASL async_b128_t* to_lds_b128(void* p) {
  // LDS flat addresses carry the 0-based LDS offset in the low 32 bits.
  return (ASL async_b128_t*)(unsigned int)(size_t)p;
}

__device__ __forceinline__ v8f wmma_bf16(v16bf a, v16bf b, v8f c) {
  // D = A(16x32 bf16) * B(32x16 bf16) + C(16x16 f32)
  return __builtin_amdgcn_wmma_f32_16x16x32_bf16(false, a, false, b, (short)0, c,
                                                 false, false);
}

__device__ __forceinline__ v16bf frag_ld(const bf16_t* p0, const bf16_t* p1) {
  union { uint4 q[2]; v16bf v; } u;
  u.q[0] = *(const uint4*)p0;   // elements 0..7
  u.q[1] = *(const uint4*)p1;   // elements 8..15
  return u.v;
}

__device__ __forceinline__ v8f vzero8() {
  v8f z = {0.f, 0.f, 0.f, 0.f, 0.f, 0.f, 0.f, 0.f};
  return z;
}

// ---------------------------------------------------------------------------
// Kernel 0: lambda = exp(lq1.lk1) - exp(lq2.lk2) + 0.8
// ---------------------------------------------------------------------------
__global__ void lambda_kernel(const float* __restrict__ lq1, const float* __restrict__ lk1,
                              const float* __restrict__ lq2, const float* __restrict__ lk2,
                              float* __restrict__ lam) {
  if (threadIdx.x == 0) {
    float d1 = 0.f, d2 = 0.f;
    for (int i = 0; i < DHEAD; i++) { d1 += lq1[i] * lk1[i]; d2 += lq2[i] * lk2[i]; }
    *lam = __expf(d1) - __expf(d2) + 0.8f;
  }
}

// ---------------------------------------------------------------------------
// Kernel 1: f32 -> bf16 elementwise (4 elems / thread)
// ---------------------------------------------------------------------------
__global__ void cvt_f32_bf16_kernel(const float* __restrict__ in, bf16_t* __restrict__ out) {
  size_t i = (size_t)blockIdx.x * blockDim.x + threadIdx.x;
  float4 f = ((const float4*)in)[i];
  v4bf b;
  b.x = (__bf16)f.x; b.y = (__bf16)f.y; b.z = (__bf16)f.z; b.w = (__bf16)f.w;
  *(v4bf*)(out + 4 * i) = b;
}

// ---------------------------------------------------------------------------
// Kernel 2: transpose + convert: in (R x C, f32) -> out (C x R, bf16)
// ---------------------------------------------------------------------------
__global__ void transpose_cvt_kernel(const float* __restrict__ in, bf16_t* __restrict__ out,
                                     int R, int C) {
  __shared__ float tile[32][33];
  const int c0 = blockIdx.x * 32, r0 = blockIdx.y * 32;
  const int tx = threadIdx.x, ty = threadIdx.y;   // 32 x 8
  #pragma unroll
  for (int i = 0; i < 32; i += 8)
    tile[ty + i][tx] = in[(size_t)(r0 + ty + i) * C + c0 + tx];
  __syncthreads();
  #pragma unroll
  for (int i = 0; i < 32; i += 8)
    out[(size_t)(c0 + ty + i) * R + r0 + tx] = (bf16_t)tile[tx][ty + i];
}

// ---------------------------------------------------------------------------
// Kernel 3: GEMM  C(MxN) = A(MxK, bf16 row-major) * Bt(NxK, bf16 row-major)^T
// Block 128x128, BK=32, 256 threads = 8 waves (4 M-waves x 2 N-waves),
// double-buffered LDS; A staged via async-to-LDS, B via TDM (when available).
// ---------------------------------------------------------------------------
constexpr int GEMM_LDT = 40;   // padded LDS row stride (bf16 units): 64B data + 16B pad

__device__ __forceinline__ void stage_tile(const bf16_t* __restrict__ A,
                                           const bf16_t* __restrict__ Bt,
                                           bf16_t* Asb, bf16_t* Bsb,
                                           int bm, int bn, int K, int kb,
                                           int lr, int lc, int wave) {
#if USE_ASYNC_COPY
  #pragma unroll
  for (int p = 0; p < 2; p++) {
    const int row = lr + p * 64;
    __builtin_amdgcn_global_load_async_to_lds_b128(
        to_global_b128(A + (size_t)(bm + row) * K + kb + lc),
        to_lds_b128(Asb + row * GEMM_LDT + lc), 0, 0);
#if !USE_TDM
    __builtin_amdgcn_global_load_async_to_lds_b128(
        to_global_b128(Bt + (size_t)(bn + row) * K + kb + lc),
        to_lds_b128(Bsb + row * GEMM_LDT + lc), 0, 0);
#endif
  }
#else
  #pragma unroll
  for (int p = 0; p < 2; p++) {
    const int row = lr + p * 64;
    *(uint4*)(Asb + row * GEMM_LDT + lc) = *(const uint4*)(A + (size_t)(bm + row) * K + kb + lc);
#if !USE_TDM
    *(uint4*)(Bsb + row * GEMM_LDT + lc) = *(const uint4*)(Bt + (size_t)(bn + row) * K + kb + lc);
#endif
  }
#endif

#if USE_TDM
  // One TDM descriptor moves the whole 128x32-element B tile (wave 0 only).
  if (wave == 0) {
    const unsigned long long gaddr =
        (unsigned long long)(size_t)(Bt + (size_t)bn * K + kb);
    const unsigned int laddr = (unsigned int)(size_t)(void*)Bsb;  // LDS byte offset
    tdm_u32x4 g0;
    g0[0] = 1u;                                    // count=1 (valid user descriptor)
    g0[1] = laddr;                                 // lds_addr
    g0[2] = (unsigned int)gaddr;                   // global_addr[31:0]
    g0[3] = (unsigned int)(gaddr >> 32) | (2u << 30);  // global_addr[56:32] | type=2
    tdm_i32x8 g1;
    // data_size=2B (code 1), pad_enable, pad_interval=3 (16 DW), pad_amount=3 (4 DW)
    g1[0] = (int)((1u << 16) | (1u << 20) | (3u << 22) | (3u << 25));
    const unsigned int td0 = (unsigned int)K;      // tensor_dim0 (elements)
    const unsigned int td1 = 128u;                 // tensor_dim1
    g1[1] = (int)((td0 & 0xFFFFu) << 16);          // tensor_dim0[15:0] @ bits 63:48
    g1[2] = (int)((td0 >> 16) | ((td1 & 0xFFFFu) << 16));
    g1[3] = (int)((td1 >> 16) | (32u << 16));      // tile_dim0 = 32 elements
    g1[4] = (int)128u;                             // tile_dim1 = 128, tile_dim2 = 0
    g1[5] = (int)(unsigned int)K;                  // tensor_dim0_stride[31:0]
    g1[6] = 0;                                     // stride0 hi16 = 0, stride1 lo = 0
    g1[7] = 0;
    tdm_i32x4 z4 = {0, 0, 0, 0};
    tdm_i32x8 z8 = {0, 0, 0, 0, 0, 0, 0, 0};
    __builtin_amdgcn_tensor_load_to_lds(g0, g1, z4, z4, z8, 0);
  }
#endif
}

__device__ __forceinline__ void compute_tile(const bf16_t* Asb, const bf16_t* Bsb,
                                             int wm, int wn, int m16, int kh,
                                             v8f acc[2][4]) {
  v16bf af[2], bfrag[4];
  #pragma unroll
  for (int mt = 0; mt < 2; mt++) {
    // A frag: lane holds row m16 of subtile; K = kh*8..+7 and kh*8+16..+23
    const bf16_t* pa = Asb + (wm * 32 + mt * 16 + m16) * GEMM_LDT + kh * 8;
    af[mt] = frag_ld(pa, pa + 16);
  }
  #pragma unroll
  for (int nt = 0; nt < 4; nt++) {
    // B frag: lane holds col m16 of subtile; K = kh*8..+7 and kh*8+16..+23
    const bf16_t* pb = Bsb + (wn * 64 + nt * 16 + m16) * GEMM_LDT + kh * 8;
    bfrag[nt] = frag_ld(pb, pb + 16);
  }
  #pragma unroll
  for (int mt = 0; mt < 2; mt++)
    #pragma unroll
    for (int nt = 0; nt < 4; nt++)
      acc[mt][nt] = wmma_bf16(af[mt], bfrag[nt], acc[mt][nt]);
}

template <bool BF16OUT>
__global__ void __launch_bounds__(256)
gemm_bf16_128x128(const bf16_t* __restrict__ A, const bf16_t* __restrict__ Bt,
                  float* __restrict__ Cf, bf16_t* __restrict__ Cb,
                  int M, int N, int K) {
  __shared__ __align__(16) bf16_t As[2][128 * GEMM_LDT];
  __shared__ __align__(16) bf16_t Bs[2][128 * GEMM_LDT];

  const int tid  = threadIdx.x;
  const int wave = tid >> 5, lane = tid & 31;
  const int wm = wave & 3, wn = wave >> 2;
  const int m16 = lane & 15, kh = lane >> 4;
  const int bm = blockIdx.y * 128, bn = blockIdx.x * 128;
  const int lr = tid >> 2, lc = (tid & 3) * 8;    // tile loader: 64 rows x 4 chunks

  v8f acc[2][4];
  #pragma unroll
  for (int i = 0; i < 2; i++)
    #pragma unroll
    for (int j = 0; j < 4; j++) acc[i][j] = vzero8();

  const int nkt = K >> 5;
  stage_tile(A, Bt, As[0], Bs[0], bm, bn, K, 0, lr, lc, wave);

  int i = 0;
  for (; i + 1 < nkt; ++i) {
    stage_tile(A, Bt, As[(i + 1) & 1], Bs[(i + 1) & 1], bm, bn, K, (i + 1) * 32,
               lr, lc, wave);
#if USE_ASYNC_COPY
    __builtin_amdgcn_s_wait_asynccnt(ASYNC_PER_THREAD);  // tile i done; i+1 in flight
#endif
#if USE_TDM
    if (wave == 0) __builtin_amdgcn_s_wait_tensorcnt(1);
#endif
    __syncthreads();
    compute_tile(As[i & 1], Bs[i & 1], wm, wn, m16, kh, acc);
    __syncthreads();
  }
#if USE_ASYNC_COPY
  __builtin_amdgcn_s_wait_asynccnt(0);
#endif
#if USE_TDM
  if (wave == 0) __builtin_amdgcn_s_wait_tensorcnt(0);
#endif
  __syncthreads();
  compute_tile(As[i & 1], Bs[i & 1], wm, wn, m16, kh, acc);

  // C/D layout: lanes 0-15 -> M = r (vgpr), lanes 16-31 -> M = 8 + r; N = m16.
  #pragma unroll
  for (int mt = 0; mt < 2; mt++)
    #pragma unroll
    for (int nt = 0; nt < 4; nt++)
      #pragma unroll
      for (int r = 0; r < 8; r++) {
        const int row = bm + wm * 32 + mt * 16 + kh * 8 + r;
        const int col = bn + wn * 64 + nt * 16 + m16;
        if constexpr (BF16OUT) Cb[(size_t)row * N + col] = (bf16_t)acc[mt][nt][r];
        else                   Cf[(size_t)row * N + col] = acc[mt][nt][r];
      }
}

// ---------------------------------------------------------------------------
// Kernel 4: fused differential flash attention.
// Grid: (SEQ/64, BS*NHEADS). Block 256 threads = 8 waves:
//   comp = wave/4 (attention component), sub = wave%4 (16-query subtile).
// Each wave: 16 queries x full dh=64, online softmax in C/D layout.
// comp1 result combined with comp0 via LDS: o = (o1 - lam*o2) * (1 - 0.8).
// ---------------------------------------------------------------------------
__global__ void __launch_bounds__(256)
diff_attn_kernel(const bf16_t* __restrict__ Q, const bf16_t* __restrict__ Kmat,
                 const bf16_t* __restrict__ V, bf16_t* __restrict__ O,
                 const float* __restrict__ lam_p) {
  __shared__ __align__(16) bf16_t Pbuf[8][16 * 64];   // per-wave P staging (16 KB)
  __shared__ __align__(16) bf16_t Vt[2][64][64];      // V transposed [comp][dh][key] (16 KB)
  __shared__ __align__(16) float  O2[64][64];         // comp1 output (16 KB)

  const int tid  = threadIdx.x;
  const int wave = tid >> 5, lane = tid & 31;
  const int comp = wave >> 2, sub = wave & 3;
  const int m16 = lane & 15, kh = lane >> 4;
  const int bq = blockIdx.x, bh = blockIdx.y;
  const int b = bh >> 4, h = bh & 15;

  const size_t hoff = (size_t)b * SEQ * 2048 + (size_t)comp * 1024 + h * 64;
  const bf16_t* Qb = Q + hoff;
  const bf16_t* Kb = Kmat + hoff;
  const int qs = bq * 64 + sub * 16;

  // Q fragments (A-layout), loaded once: dh chunks [0,32) and [32,64)
  v16bf aq[2];
  {
    const bf16_t* qr = Qb + (size_t)(qs + m16) * 2048;
    aq[0] = frag_ld(qr + kh * 8,      qr + kh * 8 + 16);
    aq[1] = frag_ld(qr + 32 + kh * 8, qr + 32 + kh * 8 + 16);
  }

  v8f o[4];
  #pragma unroll
  for (int nt = 0; nt < 4; nt++) o[nt] = vzero8();
  v8f m8, l8 = vzero8();
  #pragma unroll
  for (int r = 0; r < 8; r++) m8[r] = -3.0e38f;

  // cooperative V-tile loader mapping (256 threads cover both comp tiles)
  const int vcomp = tid >> 7;
  const int vu = tid & 127;
  const int vk = vu >> 3;            // key row 0..15 (+p)
  const int vch = (vu & 7) * 8;      // dh chunk base
  const bf16_t* Vb = V + (size_t)b * SEQ * 2048 + (size_t)vcomp * 1024 + h * 64;

  for (int kb = 0; kb <= bq; kb++) {
    const int kbase = kb * 64;

    // prefetch next tile (global_prefetch_b8)
    if (kb < bq) {
      const int nk = kbase + 64;
      __builtin_prefetch(Kb + (size_t)(nk + lane) * 2048, 0, 1);
      __builtin_prefetch(Kb + (size_t)(nk + 32 + lane) * 2048, 0, 1);
      __builtin_prefetch(Vb + (size_t)(nk + vk) * 2048 + vch, 0, 1);
      __builtin_prefetch(Vb + (size_t)(nk + 32 + vk) * 2048 + vch, 0, 1);
    }

    __syncthreads();                                   // protect Vt overwrite
    #pragma unroll
    for (int p = 0; p < 64; p += 16) {
      union { uint4 q; bf16_t e[8]; } u;
      u.q = *(const uint4*)(Vb + (size_t)(kbase + vk + p) * 2048 + vch);
      #pragma unroll
      for (int j = 0; j < 8; j++) Vt[vcomp][vch + j][vk + p] = u.e[j];
    }
    __syncthreads();

    // S = (Q K^T) * scale ; B frag = K row (key m16), dh kh*8..+7 and kh*8+16..+23
    v8f s[4];
    #pragma unroll
    for (int nt = 0; nt < 4; nt++) {
      const bf16_t* kr = Kb + (size_t)(kbase + nt * 16 + m16) * 2048;
      v16bf b0 = frag_ld(kr + kh * 8,      kr + kh * 8 + 16);
      v16bf b1 = frag_ld(kr + 32 + kh * 8, kr + 32 + kh * 8 + 16);
      v8f t = vzero8();
      t = wmma_bf16(aq[0], b0, t);
      t = wmma_bf16(aq[1], b1, t);
      s[nt] = t * 0.125f;                              // 1/sqrt(64)
    }

    if (kb == bq) {                                    // causal mask, diagonal block
      #pragma unroll
      for (int nt = 0; nt < 4; nt++) {
        const int kidx = kbase + nt * 16 + m16;
        #pragma unroll
        for (int r = 0; r < 8; r++)
          if (kidx > qs + kh * 8 + r) s[nt][r] = -3.0e38f;
      }
    }

    // online softmax: row stats live per accumulator slot, reduce over 16 lanes
    v8f tmax = s[0];
    #pragma unroll
    for (int nt = 1; nt < 4; nt++)
      #pragma unroll
      for (int r = 0; r < 8; r++) tmax[r] = fmaxf(tmax[r], s[nt][r]);
    #pragma unroll
    for (int off = 1; off < 16; off <<= 1)
      #pragma unroll
      for (int r = 0; r < 8; r++) tmax[r] = fmaxf(tmax[r], __shfl_xor(tmax[r], off));

    v8f corr;
    #pragma unroll
    for (int r = 0; r < 8; r++) {
      const float mn = fmaxf(m8[r], tmax[r]);
      corr[r] = __expf(m8[r] - mn);
      m8[r] = mn;
    }
    l8 = l8 * corr;
    #pragma unroll
    for (int nt = 0; nt < 4; nt++) o[nt] = o[nt] * corr;

    v8f ps = vzero8();
    #pragma unroll
    for (int nt = 0; nt < 4; nt++)
      #pragma unroll
      for (int r = 0; r < 8; r++) {
        const float e = __expf(s[nt][r] - m8[r]);
        s[nt][r] = e;
        ps[r] += e;
      }
    #pragma unroll
    for (int off = 1; off < 16; off <<= 1)
      #pragma unroll
      for (int r = 0; r < 8; r++) ps[r] += __shfl_xor(ps[r], off);
    l8 = l8 + ps;

    // P: C/D layout -> LDS -> A layout (in-wave LDS ops are in-order)
    bf16_t* pw = Pbuf[wave];
    #pragma unroll
    for (int nt = 0; nt < 4; nt++)
      #pragma unroll
      for (int r = 0; r < 8; r++)
        pw[(kh * 8 + r) * 64 + nt * 16 + m16] = (bf16_t)s[nt][r];

    const bf16_t* pr = Pbuf[wave] + m16 * 64;
    v16bf ap0 = frag_ld(pr + kh * 8,      pr + kh * 8 + 16);        // keys [0,32)
    v16bf ap1 = frag_ld(pr + 32 + kh * 8, pr + 32 + kh * 8 + 16);   // keys [32,64)

    // O += P * V ; B frag = Vt row (dh m16), keys kh*8..+7 and kh*8+16..+23
    #pragma unroll
    for (int nt = 0; nt < 4; nt++) {
      const bf16_t* vr = &Vt[comp][nt * 16 + m16][0];
      v16bf bv0 = frag_ld(vr + kh * 8,      vr + kh * 8 + 16);
      v16bf bv1 = frag_ld(vr + 32 + kh * 8, vr + 32 + kh * 8 + 16);
      o[nt] = wmma_bf16(ap0, bv0, o[nt]);
      o[nt] = wmma_bf16(ap1, bv1, o[nt]);
    }
  }

  #pragma unroll
  for (int nt = 0; nt < 4; nt++) o[nt] = o[nt] / l8;

  if (comp == 1) {
    #pragma unroll
    for (int nt = 0; nt < 4; nt++)
      #pragma unroll
      for (int r = 0; r < 8; r++)
        O2[sub * 16 + kh * 8 + r][nt * 16 + m16] = o[nt][r];
  }
  __syncthreads();
  if (comp == 0) {
    const float lam = *lam_p;
    const float sc = 1.0f - 0.8f;
    #pragma unroll
    for (int nt = 0; nt < 4; nt++)
      #pragma unroll
      for (int r = 0; r < 8; r++) {
        const float v2 = O2[sub * 16 + kh * 8 + r][nt * 16 + m16];
        const float res = (o[nt][r] - lam * v2) * sc;
        const int row = b * SEQ + bq * 64 + sub * 16 + kh * 8 + r;
        O[(size_t)row * 1024 + h * 64 + nt * 16 + m16] = (bf16_t)res;
      }
  }
}

// ---------------------------------------------------------------------------
// Host orchestration
// ---------------------------------------------------------------------------
extern "C" void kernel_launch(void* const* d_in, const int* in_sizes, int n_in,
                              void* d_out, int out_size, void* d_ws, size_t ws_size,
                              hipStream_t stream) {
  (void)in_sizes; (void)n_in; (void)out_size; (void)ws_size;
  const float* x   = (const float*)d_in[0];
  const float* Wq  = (const float*)d_in[1];
  const float* Wk  = (const float*)d_in[2];
  const float* Wv  = (const float*)d_in[3];
  const float* Wo  = (const float*)d_in[4];
  const float* lq1 = (const float*)d_in[5];
  const float* lk1 = (const float*)d_in[6];
  const float* lq2 = (const float*)d_in[7];
  const float* lk2 = (const float*)d_in[8];
  float* out = (float*)d_out;

  char* ws = (char*)d_ws;
  size_t off = 0;
  auto alloc = [&](size_t bytes) -> void* {
    void* p = ws + off;
    off += (bytes + 255) & ~(size_t)255;
    return p;
  };
  float*  lam = (float*)alloc(sizeof(float));
  bf16_t* xb  = (bf16_t*)alloc((size_t)4096 * 1024 * 2);
  bf16_t* WqT = (bf16_t*)alloc((size_t)2048 * 1024 * 2);
  bf16_t* WkT = (bf16_t*)alloc((size_t)2048 * 1024 * 2);
  bf16_t* WvT = (bf16_t*)alloc((size_t)2048 * 1024 * 2);
  bf16_t* WoT = (bf16_t*)alloc((size_t)1024 * 1024 * 2);
  bf16_t* Qm  = (bf16_t*)alloc((size_t)4096 * 2048 * 2);
  bf16_t* Km  = (bf16_t*)alloc((size_t)4096 * 2048 * 2);
  bf16_t* Vm  = (bf16_t*)alloc((size_t)4096 * 2048 * 2);
  bf16_t* Om  = (bf16_t*)alloc((size_t)4096 * 1024 * 2);

  lambda_kernel<<<1, 32, 0, stream>>>(lq1, lk1, lq2, lk2, lam);

  cvt_f32_bf16_kernel<<<4096, 256, 0, stream>>>(x, xb);    // 4096*1024 elems / 4

  dim3 tb(32, 8);
  transpose_cvt_kernel<<<dim3(2048 / 32, 1024 / 32), tb, 0, stream>>>(Wq, WqT, 1024, 2048);
  transpose_cvt_kernel<<<dim3(2048 / 32, 1024 / 32), tb, 0, stream>>>(Wk, WkT, 1024, 2048);
  transpose_cvt_kernel<<<dim3(2048 / 32, 1024 / 32), tb, 0, stream>>>(Wv, WvT, 1024, 2048);
  transpose_cvt_kernel<<<dim3(1024 / 32, 1024 / 32), tb, 0, stream>>>(Wo, WoT, 1024, 1024);

  dim3 gq(2048 / 128, 4096 / 128);
  gemm_bf16_128x128<true><<<gq, 256, 0, stream>>>(xb, WqT, nullptr, Qm, 4096, 2048, 1024);
  gemm_bf16_128x128<true><<<gq, 256, 0, stream>>>(xb, WkT, nullptr, Km, 4096, 2048, 1024);
  gemm_bf16_128x128<true><<<gq, 256, 0, stream>>>(xb, WvT, nullptr, Vm, 4096, 2048, 1024);

  diff_attn_kernel<<<dim3(SEQ / 64, BS * NHEADS), 256, 0, stream>>>(Qm, Km, Vm, Om, lam);

  gemm_bf16_128x128<false><<<dim3(1024 / 128, 4096 / 128), 256, 0, stream>>>(
      Om, WoT, out, nullptr, 4096, 1024, 1024);
}
